// GNN_node_Virtualnode_30279519437415
// MI455X (gfx1250) — matrix operand order repaired and checked by reading, compiled, not run on hardware
//
#include <hip/hip_runtime.h>

// ---------------------------------------------------------------------------
// CDNA5 (gfx1250) implementation of the hypergraph-GNN forward pass.
// Dense MLPs use v_wmma_f32_16x16x32_f16 (f16 in, f32 accum).
// Assumed WMMA register layouts (CDNA5 ISA 7.12.2):
//   A (16x32 f16): lane L holds row M=L&15; element e holds
//                  K = (e&7) + ((e>>3)<<4) + ((L>>4)<<3)
//   B (32x16 f16): lane L holds column N=L&15; element e holds the same K as in A
//   C/D (16x16 f32): lane L, slot j -> (M = j + ((L>>4)<<3), N = L&15)
// Hidden activations between the two GEMMs are staged in LDS in an
// INTERLEAVED layout (columns c and c+16 adjacent) so the C->A relayout is
// one packed ds_store_b32 per 2 values and one contiguous 32B ds load per
// lane; the induced K-permutation is folded into W2 at weight-convert time.
// ---------------------------------------------------------------------------

typedef __attribute__((ext_vector_type(16))) _Float16 v16h;
typedef __attribute__((ext_vector_type(2)))  _Float16 v2h;
typedef __attribute__((ext_vector_type(8)))  float    v8f;

__device__ __forceinline__ float lrelu_f(float v) { return fmaxf(v, 0.1f * v); }

// ---------------- weight f32 -> f16 (with K zero-padding), GEMM1 W ----------
__global__ __launch_bounds__(256) void cvt_w_kernel(const float* __restrict__ src,
    _Float16* __restrict__ dst, int K, int KP, int N)
{
  int i = blockIdx.x * 256 + threadIdx.x;
  if (i >= KP * N) return;
  int k = i / N;
  dst[i] = (k < K) ? (_Float16)src[i] : (_Float16)0.f;
}

// ---------------- GEMM2 weight convert with per-32-row K permutation --------
// dst[k,n] = src[perm(k), n];  perm matches the interleaved hidden staging.
__global__ __launch_bounds__(256) void cvt_w2p_kernel(const float* __restrict__ src,
    _Float16* __restrict__ dst, int K, int N)
{
  int i = blockIdx.x * 256 + threadIdx.x;
  if (i >= K * N) return;
  int k = i / N, n = i - k * N;
  int t = k & 31;
  int g = t >> 3;
  int base = (g == 0) ? 0 : (g == 1) ? 8 : (g == 2) ? 4 : 12;
  int srck = (k & ~31) + base + ((t & 7) >> 1) + ((t & 1) << 4);
  dst[i] = (_Float16)src[(size_t)srck * N + n];
}

// ---------------- small elementwise kernels ---------------------------------
__global__ __launch_bounds__(256) void vt_kernel(const float* __restrict__ vn,
    const float* __restrict__ top, const int* __restrict__ top_batch,
    float* __restrict__ vt)
{
  int i = blockIdx.x * 256 + threadIdx.x;
  if (i < 64 * 64) {
    int v = i >> 6, c = i & 63;
    vt[i] = vn[i] + top[top_batch[v] * 64 + c];
  }
}

__global__ __launch_bounds__(256) void hin_kernel(const float* __restrict__ h,
    const float* __restrict__ vt, const int* __restrict__ batch,
    float* __restrict__ h_in, int n)
{
  int i = blockIdx.x * 256 + threadIdx.x;
  if (i < n) h_in[i] = h[i] + vt[(size_t)batch[i >> 6] * 64 + (i & 63)];
}

// ---------------- row scatter-add (segment sums over edges) -----------------
__global__ __launch_bounds__(256) void scatter_add_kernel(
    const float* __restrict__ src, const int* __restrict__ src_idx,
    const int* __restrict__ dst_idx, float* __restrict__ dst, int nE)
{
  int t = blockIdx.x * 256 + threadIdx.x;
  int e = t >> 4;
  if (e >= nE) return;
  int part = (t & 15) * 4;
  const float4 v = *(const float4*)(src + (size_t)src_idx[e] * 64 + part);
  float* d = dst + (size_t)dst_idx[e] * 64 + part;
  unsafeAtomicAdd(d + 0, v.x);
  unsafeAtomicAdd(d + 1, v.y);
  unsafeAtomicAdd(d + 2, v.z);
  unsafeAtomicAdd(d + 3, v.w);
}

// ---------------- fused LayerNorm + LeakyReLU: x = lrelu(ln(x + s*other)) ---
__global__ __launch_bounds__(256) void ln_lrelu_kernel(float* __restrict__ x,
    const float* __restrict__ other, float s, const float* __restrict__ g,
    const float* __restrict__ b, int M)
{
  int wave = threadIdx.x >> 5, lane = threadIdx.x & 31;
  int row = blockIdx.x * 8 + wave;
  if (row >= M) return;
  size_t base = (size_t)row * 64;
  float v0 = x[base + lane] + s * other[base + lane];
  float v1 = x[base + 32 + lane] + s * other[base + 32 + lane];
  float sum = v0 + v1;
  #pragma unroll
  for (int off = 16; off > 0; off >>= 1) sum += __shfl_xor(sum, off, 32);
  float mean = sum * 0.015625f;
  float d0 = v0 - mean, d1 = v1 - mean;
  float var = d0 * d0 + d1 * d1;
  #pragma unroll
  for (int off = 16; off > 0; off >>= 1) var += __shfl_xor(var, off, 32);
  float rstd = rsqrtf(var * 0.015625f + 1e-5f);
  float y0 = d0 * rstd * g[lane] + b[lane];
  float y1 = d1 * rstd * g[lane + 32] + b[lane + 32];
  x[base + lane] = lrelu_f(y0);
  x[base + 32 + lane] = lrelu_f(y1);
}

// ---------------- LDS-staged segment sum of h_in over batch -> 64 VNs -------
__global__ __launch_bounds__(256) void seg_vn_kernel(const float* __restrict__ h_in,
    const int* __restrict__ batch, float* __restrict__ vn_seg,
    float* __restrict__ cnt, int M)
{
  __shared__ float acc[64 * 64];
  __shared__ float c[64];
  const int t = threadIdx.x;
  for (int i = t; i < 4096; i += 256) acc[i] = 0.f;
  if (t < 64) c[t] = 0.f;
  __syncthreads();
  const int col = t & 63;
  const int rg = t >> 6;
  const int r0 = blockIdx.x * 1024;
  const int rEnd = (r0 + 1024 < M) ? r0 + 1024 : M;
  for (int r = r0 + rg; r < rEnd; r += 4) {
    const int v = batch[r];
    atomicAdd(&acc[v * 64 + col], h_in[(size_t)r * 64 + col]);
    if (col == 0) atomicAdd(&c[v], 1.f);
  }
  __syncthreads();
  for (int i = t; i < 4096; i += 256) unsafeAtomicAdd(&vn_seg[i], acc[i]);
  if (t < 64) unsafeAtomicAdd(&cnt[t], c[t]);
}

// ---------------- virtual-node + top-node MLP update (tiny, 1 block) --------
__global__ __launch_bounds__(256) void vn_top_update_kernel(
    const float* __restrict__ vn_seg, const float* __restrict__ cnt,
    float* __restrict__ vn, float* __restrict__ top,
    const float* __restrict__ w1, const float* __restrict__ b1,   // 64x128,128
    const float* __restrict__ w2, const float* __restrict__ b2,   // 128x64,64
    const float* __restrict__ tw, const float* __restrict__ tb,   // 64x64,64
    const int* __restrict__ top_batch)
{
  __shared__ float tmp[64 * 64];
  __shared__ float hid[64 * 128];
  __shared__ float ttmp[4 * 64];
  const int t = threadIdx.x;
  for (int i = t; i < 4096; i += 256) {
    float cc = cnt[i >> 6]; cc = cc < 1.f ? 1.f : cc;
    tmp[i] = vn_seg[i] / cc + vn[i];
  }
  __syncthreads();
  for (int i = t; i < 8192; i += 256) {
    int r = i >> 7, c = i & 127;
    float s = b1[c];
    for (int k = 0; k < 64; ++k) s += tmp[r * 64 + k] * w1[k * 128 + c];
    hid[i] = lrelu_f(s);
  }
  __syncthreads();
  for (int i = t; i < 4096; i += 256) {
    int r = i >> 6, c = i & 63;
    float s = b2[c];
    for (int k = 0; k < 128; ++k) s += hid[r * 128 + k] * w2[k * 64 + c];
    tmp[i] = vn[i] + lrelu_f(s);   // vn_new (kept in shared, reuse tmp)
  }
  __syncthreads();
  for (int i = t; i < 4096; i += 256) vn[i] = tmp[i];
  if (t < 256) {
    int tt = t >> 6, c = t & 63;
    float s = 0.f, cc = 0.f;
    for (int v = 0; v < 64; ++v)
      if (top_batch[v] == tt) { s += tmp[v * 64 + c]; cc += 1.f; }
    cc = cc < 1.f ? 1.f : cc;
    ttmp[t] = s / cc + top[t];
  }
  __syncthreads();
  if (t < 256) {
    int tt = t >> 6, c = t & 63;
    float s = tb[c];
    for (int k = 0; k < 64; ++k) s += ttmp[tt * 64 + k] * tw[k * 64 + c];
    top[t] += lrelu_f(s);
  }
}

// ---------------- fused 2-layer MLP via WMMA --------------------------------
// Out[M,DOUT] = act2( act1(A[M,DIN] @ W1 + b1) @ W2p + b2 ) (+ OtherAdd)
// One wave per 16-row tile, 8 waves/block (256 threads, wave32).
// Hidden streamed 32 cols/iter through per-wave LDS staging (interleaved).
template<int DIN, int DH, int DOUT, bool CONCAT, int ACT_MID, int ACT_OUT, bool ADD_OTHER>
__global__ __launch_bounds__(256) void fused_mlp_kernel(
    const float* __restrict__ A0, const float* __restrict__ A1,
    const float* __restrict__ A2,
    const _Float16* __restrict__ W1h, const float* __restrict__ B1,
    const _Float16* __restrict__ W2h, const float* __restrict__ B2,
    const float* __restrict__ OtherAdd, float* __restrict__ Out, int M)
{
  constexpr int LDA = CONCAT ? 64 : DIN;
  constexpr int KVALID = CONCAT ? 64 : DIN;       // valid K per source segment
  constexpr int DK1P = (DIN < 32) ? 32 : DIN;     // K padded to 32
  constexpr int KT1 = DK1P / 32;
  constexpr int NT2 = DOUT / 16;

  __shared__ _Float16 hidStage[8][16 * 32];

  const int lane = threadIdx.x & 31;
  const int wave = threadIdx.x >> 5;
  const int row0 = (blockIdx.x * 8 + wave) * 16;
  if (row0 >= M) return;
  const int rowsLeft = M - row0;
  const int m = lane & 15;
  const int khi = (lane >> 4) << 3;
  const bool rowOK = (m < rowsLeft);
  const int q = lane & 15;
  // interleaved half-position of the column pair (q, q+16) within a row
  const int p0 = ((q >> 3) << 4) + ((q & 7) << 1);

  // ---- load & convert A fragments (f32 global -> f16 WMMA A layout) ----
  v16h afrag[KT1];
  #pragma unroll
  for (int kt = 0; kt < KT1; ++kt) {
    const int kBase = kt * 32;
    const float* src = A0;
    int kb = kBase;
    if (CONCAT) {
      src = (kBase < 64) ? A0 : ((kBase < 128) ? A1 : A2);
      kb = kBase & 63;
    }
    const float* rowp = src + (size_t)(row0 + m) * LDA;
    float4 x0 = make_float4(0.f, 0.f, 0.f, 0.f), x1 = x0, x2 = x0, x3 = x0;
    const int c0 = kb + khi;
    if (rowOK) {
      if (c0 < KVALID)      { const float4* p = (const float4*)(rowp + c0);      x0 = p[0]; x1 = p[1]; }
      if (c0 + 16 < KVALID) { const float4* p = (const float4*)(rowp + c0 + 16); x2 = p[0]; x3 = p[1]; }
    }
    v16h a;
    a[0]=(_Float16)x0.x;  a[1]=(_Float16)x0.y;  a[2]=(_Float16)x0.z;  a[3]=(_Float16)x0.w;
    a[4]=(_Float16)x1.x;  a[5]=(_Float16)x1.y;  a[6]=(_Float16)x1.z;  a[7]=(_Float16)x1.w;
    a[8]=(_Float16)x2.x;  a[9]=(_Float16)x2.y;  a[10]=(_Float16)x2.z; a[11]=(_Float16)x2.w;
    a[12]=(_Float16)x3.x; a[13]=(_Float16)x3.y; a[14]=(_Float16)x3.z; a[15]=(_Float16)x3.w;
    afrag[kt] = a;
  }

  v8f oacc[NT2] = {};
  _Float16* hid = &hidStage[wave][0];

  #pragma unroll
  for (int ko = 0; ko < DH / 32; ++ko) {
    // GEMM1: produce 32 hidden columns (two 16-col tiles)
    v8f acc0 = {}, acc1 = {};
    #pragma unroll
    for (int kt = 0; kt < KT1; ++kt) {
      v16h bfr;
      for (int s = 0; s < 16; ++s) bfr[s] = W1h[(size_t)(kt * 32 + khi + (s & 7) + ((s >> 3) << 4)) * DH + ko * 32 + q];
      acc0 = __builtin_amdgcn_wmma_f32_16x16x32_f16(
          false, afrag[kt], false, bfr, (short)0, acc0, false, false);
    }
    #pragma unroll
    for (int kt = 0; kt < KT1; ++kt) {
      v16h bfr;
      for (int s = 0; s < 16; ++s) bfr[s] = W1h[(size_t)(kt * 32 + khi + (s & 7) + ((s >> 3) << 4)) * DH + ko * 32 + 16 + q];
      acc1 = __builtin_amdgcn_wmma_f32_16x16x32_f16(
          false, afrag[kt], false, bfr, (short)0, acc1, false, false);
    }
    // epilogue 1: bias + act, pack column pair (q, q+16) -> one b32 DS store
    const float bias0 = B1[ko * 32 + q];
    const float bias1 = B1[ko * 32 + 16 + q];
    #pragma unroll
    for (int j = 0; j < 8; ++j) {
      float v0 = acc0[j] + bias0;
      float v1 = acc1[j] + bias1;
      if (ACT_MID == 1) { v0 = lrelu_f(v0); v1 = lrelu_f(v1); }
      else              { v0 = fmaxf(v0, 0.f); v1 = fmaxf(v1, 0.f); }
      v2h pk; pk[0] = (_Float16)v0; pk[1] = (_Float16)v1;
      *(v2h*)(hid + (j + khi) * 32 + p0) = pk;
    }
    // Re-load the 16x32 hidden chunk as a contiguous 32B A-fragment per lane
    // (interleaved K order is folded into W2h's row permutation).
    v16h hf = *(const v16h*)(hid + m * 32 + (khi << 1));
    // GEMM2: accumulate into output tiles
    #pragma unroll
    for (int ot = 0; ot < NT2; ++ot) {
      v16h bfr;
      for (int s = 0; s < 16; ++s) bfr[s] = W2h[(size_t)(ko * 32 + khi + (s & 7) + ((s >> 3) << 4)) * DOUT + ot * 16 + q];
      oacc[ot] = __builtin_amdgcn_wmma_f32_16x16x32_f16(
          false, hf, false, bfr, (short)0, oacc[ot], false, false);
    }
  }

  // ---- epilogue: bias, activation, optional residual add, store -----------
  #pragma unroll
  for (int ot = 0; ot < NT2; ++ot) {
    const int col = ot * 16 + q;
    const float bias = B2[col];
    #pragma unroll
    for (int j = 0; j < 8; ++j) {
      const int r = j + khi;
      if (r < rowsLeft) {
        const size_t off = (size_t)(row0 + r) * DOUT + col;
        float v = oacc[ot][j] + bias;
        if (ACT_OUT == 1) v = lrelu_f(v);
        if (ADD_OTHER) v += OtherAdd[off];
        Out[off] = v;
      }
    }
  }
}

// ---------------------------------------------------------------------------
extern "C" void kernel_launch(void* const* d_in, const int* in_sizes, int n_in,
                              void* d_out, int out_size, void* d_ws, size_t ws_size,
                              hipStream_t stream)
{
  (void)n_in; (void)out_size; (void)ws_size;
  const float* x        = (const float*)d_in[0];
  const float* x_net    = (const float*)d_in[1];
  const int* pin_inst   = (const int*)d_in[2];
  const int* pin_net    = (const int*)d_in[3];
  const int* drive_inst = (const int*)d_in[4];
  const int* drive_net  = (const int*)d_in[5];
  const int* sink_inst  = (const int*)d_in[6];
  const int* sink_net   = (const int*)d_in[7];
  const int* batch      = (const int*)d_in[8];
  const int* top_batch  = (const int*)d_in[9];
  const int nInst  = in_sizes[0] / 16;   // 400000
  const int nNet   = in_sizes[1] / 16;   // 300000
  const int nPin   = in_sizes[2];
  const int nDrive = in_sizes[4];
  const int nSink  = in_sizes[6];

  // Param leaf order: top-level dict insertion order; nested dicts sorted-key.
  //  10: enc.b1(128) 11: enc.b2(64) 12: enc.w1(16x128) 13: enc.w2(128x64)
  //  14: enc_net.b1  15: enc_net.b2 16: enc_net.w1(16x64) 17: enc_net.w2(64x64)
  //  layer base (l0=18, l1=38, l2=58):
  //   +0 ln_b +1 ln_g +2 mlp.b1 +3 mlp.b2 +4 mlp.w1(192x192) +5 mlp.w2(192x64)
  //   +6 phi.b1 +7 phi.b2 +8 phi.w1 +9 phi.w2
  //   +10 psi.b1 +11 psi.b2 +12 psi.w1 +13 psi.w2
  //   (l<2) +14 top_mlp.b +15 top_mlp.w +16 vn.b1 +17 vn.b2 +18 vn.w1 +19 vn.w2
  auto P = [&](int i) { return (const float*)d_in[i]; };

  // ---- workspace bump allocator ----
  uintptr_t cur = (uintptr_t)d_ws;
  auto alloc = [&](size_t bytes) {
    uintptr_t p = (cur + 255) & ~(uintptr_t)255; cur = p + bytes; return (void*)p;
  };
  float* h       = (float*)alloc((size_t)nInst * 64 * 4);
  float* h_in    = (float*)alloc((size_t)nInst * 64 * 4);
  float* h_drive = (float*)alloc((size_t)nInst * 64 * 4);
  float* sinkb   = (float*)alloc((size_t)nInst * 64 * 4);   // sink_acc -> h_sink
  float* pinb    = (float*)alloc((size_t)nNet  * 64 * 4);   // pin_acc  -> net_agg
  float* hn      = (float*)d_out;                           // net features live in d_out
  float* vn      = (float*)alloc(64 * 64 * 4);
  float* top     = (float*)alloc(4 * 64 * 4);
  float* vt      = (float*)alloc(64 * 64 * 4);
  float* vn_seg  = (float*)alloc(64 * 64 * 4);
  float* cnt     = (float*)alloc(64 * 4);

  _Float16* encW1h  = (_Float16*)alloc(32 * 128 * 2);
  _Float16* encW2h  = (_Float16*)alloc(128 * 64 * 2);
  _Float16* encnW1h = (_Float16*)alloc(32 * 64 * 2);
  _Float16* encnW2h = (_Float16*)alloc(64 * 64 * 2);
  _Float16 *phiW1h[3], *phiW2h[3], *psiW1h[3], *psiW2h[3], *mlpW1h[3], *mlpW2h[3];
  for (int l = 0; l < 3; ++l) {
    phiW1h[l] = (_Float16*)alloc(64 * 64 * 2);
    phiW2h[l] = (_Float16*)alloc(64 * 64 * 2);
    psiW1h[l] = (_Float16*)alloc(64 * 64 * 2);
    psiW2h[l] = (_Float16*)alloc(64 * 64 * 2);
    mlpW1h[l] = (_Float16*)alloc(192 * 192 * 2);
    mlpW2h[l] = (_Float16*)alloc(192 * 64 * 2);
  }

  auto cvt1 = [&](const float* src, _Float16* dst, int K, int KP, int N) {
    int n = KP * N;
    cvt_w_kernel<<<(n + 255) / 256, 256, 0, stream>>>(src, dst, K, KP, N);
  };
  auto cvt2 = [&](const float* src, _Float16* dst, int K, int N) {
    int n = K * N;
    cvt_w2p_kernel<<<(n + 255) / 256, 256, 0, stream>>>(src, dst, K, N);
  };
  cvt1(P(12), encW1h, 16, 32, 128);
  cvt2(P(13), encW2h, 128, 64);
  cvt1(P(16), encnW1h, 16, 32, 64);
  cvt2(P(17), encnW2h, 64, 64);
  for (int l = 0; l < 3; ++l) {
    const int base = (l < 2) ? 18 + 20 * l : 58;
    cvt1(P(base + 4), mlpW1h[l], 192, 192, 192);
    cvt2(P(base + 5), mlpW2h[l], 192, 64);
    cvt1(P(base + 8), phiW1h[l], 64, 64, 64);
    cvt2(P(base + 9), phiW2h[l], 64, 64);
    cvt1(P(base + 12), psiW1h[l], 64, 64, 64);
    cvt2(P(base + 13), psiW2h[l], 64, 64);
  }

  const int gInst = (nInst + 127) / 128;
  const int gNet  = (nNet + 127) / 128;

  // ---- encoders (mlp_lrelu): h = enc(x), hn = enc_net(x_net) ----
  fused_mlp_kernel<16, 128, 64, false, 1, 1, false><<<gInst, 256, 0, stream>>>(
      x, nullptr, nullptr, encW1h, P(10), encW2h, P(11), nullptr, h, nInst);
  fused_mlp_kernel<16, 64, 64, false, 1, 1, false><<<gNet, 256, 0, stream>>>(
      x_net, nullptr, nullptr, encnW1h, P(14), encnW2h, P(15), nullptr, hn, nNet);

  hipMemsetAsync(vn, 0, 64 * 64 * 4, stream);   // vn0 == 0
  hipMemsetAsync(top, 0, 4 * 64 * 4, stream);   // top0 == 0

  for (int l = 0; l < 3; ++l) {
    const int base = (l < 2) ? 18 + 20 * l : 58;
    const float* ln_b = P(base + 0);
    const float* ln_g = P(base + 1);

    // h_in = h + (vn + top[top_batch])[batch]
    vt_kernel<<<16, 256, 0, stream>>>(vn, top, top_batch, vt);
    hin_kernel<<<(nInst * 64 + 255) / 256, 256, 0, stream>>>(h, vt, batch, h_in, nInst * 64);

    // net_agg = relu-MLP(phi)(segsum(h_in over pins)) + hn
    hipMemsetAsync(pinb, 0, (size_t)nNet * 64 * 4, stream);
    scatter_add_kernel<<<(nPin * 16 + 255) / 256, 256, 0, stream>>>(
        h_in, pin_inst, pin_net, pinb, nPin);
    fused_mlp_kernel<64, 64, 64, false, 0, 0, true><<<gNet, 256, 0, stream>>>(
        pinb, nullptr, nullptr, phiW1h[l], P(base + 6), phiW2h[l], P(base + 7),
        hn, pinb, nNet);

    // h_drive / h_sink segment sums from net_agg
    hipMemsetAsync(h_drive, 0, (size_t)nInst * 64 * 4, stream);
    hipMemsetAsync(sinkb, 0, (size_t)nInst * 64 * 4, stream);
    scatter_add_kernel<<<(nDrive * 16 + 255) / 256, 256, 0, stream>>>(
        pinb, drive_net, drive_inst, h_drive, nDrive);
    scatter_add_kernel<<<(nSink * 16 + 255) / 256, 256, 0, stream>>>(
        pinb, sink_net, sink_inst, sinkb, nSink);
    // h_sink = psi(sink_acc), in-place
    fused_mlp_kernel<64, 64, 64, false, 0, 0, false><<<gInst, 256, 0, stream>>>(
        sinkb, nullptr, nullptr, psiW1h[l], P(base + 10), psiW2h[l], P(base + 11),
        nullptr, sinkb, nInst);
    // mlp_out = mlp([h_in | h_drive | h_sink]) -> h
    fused_mlp_kernel<192, 192, 64, true, 0, 0, false><<<gInst, 256, 0, stream>>>(
        h_in, h_drive, sinkb, mlpW1h[l], P(base + 2), mlpW2h[l], P(base + 3),
        nullptr, h, nInst);

    // virtual-node updates (use h_in and old vn/top) for layers 0,1
    if (l < 2) {
      hipMemsetAsync(vn_seg, 0, 64 * 64 * 4, stream);
      hipMemsetAsync(cnt, 0, 64 * 4, stream);
      seg_vn_kernel<<<(nInst + 1023) / 1024, 256, 0, stream>>>(h_in, batch, vn_seg, cnt, nInst);
      vn_top_update_kernel<<<1, 256, 0, stream>>>(
          vn_seg, cnt, vn, top,
          P(base + 18), P(base + 16), P(base + 19), P(base + 17),
          P(base + 15), P(base + 14), top_batch);
    }

    // h  = lrelu(ln(mlp_out + 2*h_in));  hn = lrelu(ln(net_agg + hn))
    ln_lrelu_kernel<<<(nInst + 7) / 8, 256, 0, stream>>>(h, h_in, 2.0f, ln_g, ln_b, nInst);
    ln_lrelu_kernel<<<(nNet + 7) / 8, 256, 0, stream>>>(hn, pinb, 1.0f, ln_g, ln_b, nNet);
  }
  // final hn already lives in d_out
}
